// ASNST_Layer_30528627540540
// MI455X (gfx1250) — compile-verified, matrix operand order checked by reading, not run on hardware
//
#include <hip/hip_runtime.h>

// ---------------------------------------------------------------------------
// Types / helpers
// ---------------------------------------------------------------------------
typedef __attribute__((ext_vector_type(16))) __bf16        v16bf;
typedef __attribute__((ext_vector_type(8)))  float         v8f;
typedef __attribute__((ext_vector_type(8)))  unsigned int  v8u;

__device__ __forceinline__ unsigned short f2bf(float f) {
    unsigned int u = __builtin_bit_cast(unsigned int, f);
    u = u + 0x7FFFu + ((u >> 16) & 1u);   // round-to-nearest-even
    return (unsigned short)(u >> 16);
}

__device__ __forceinline__ v8f wmma_bf16(v8u a, v8u b, v8f c) {
    return __builtin_amdgcn_wmma_f32_16x16x32_bf16(
        /*neg_a=*/false, __builtin_bit_cast(v16bf, a),
        /*neg_b=*/false, __builtin_bit_cast(v16bf, b),
        /*c_mod=*/(short)0, c, /*reuse_a=*/false, /*reuse_b=*/false);
}

// Async global -> LDS copy of 16 bytes (CDNA5, tracked by ASYNCcnt).
__device__ __forceinline__ void async_g2l_b128(void* lds_ptr, const void* gptr) {
    unsigned lds_off = (unsigned)(unsigned long long)lds_ptr;  // addr[31:0] = LDS offset
    asm volatile("global_load_async_to_lds_b128 %0, %1, off"
                 :
                 : "v"(lds_off), "v"(gptr)
                 : "memory");
}

__device__ __forceinline__ void wait_async0() {
    asm volatile("s_wait_asynccnt 0x0" ::: "memory");
}

// ---------------------------------------------------------------------------
// fp32 -> bf16 conversion (weights)
// ---------------------------------------------------------------------------
__global__ void cvt_f32_bf16(const float* __restrict__ in,
                             unsigned short* __restrict__ out, long n) {
    long i  = (long)blockIdx.x * blockDim.x + threadIdx.x;
    long st = (long)gridDim.x * blockDim.x;
    for (; i < n; i += st) out[i] = f2bf(in[i]);
}

// ---------------------------------------------------------------------------
// LayerNorm (C=1024), output bf16.  One row per 256-thread block.
// ---------------------------------------------------------------------------
__global__ __launch_bounds__(256)
void ln_bf16_kernel(const float* __restrict__ x, const float* __restrict__ g,
                    const float* __restrict__ b, unsigned short* __restrict__ y) {
    __shared__ float rs1[256];
    __shared__ float rs2[256];
    const int  row = blockIdx.x;
    const float* xr = x + (long)row * 1024;
    float v[4];
    float s = 0.f, q = 0.f;
#pragma unroll
    for (int i = 0; i < 4; ++i) {
        v[i] = xr[threadIdx.x + 256 * i];
        s += v[i];
        q += v[i] * v[i];
    }
    rs1[threadIdx.x] = s; rs2[threadIdx.x] = q;
    __syncthreads();
    for (int st = 128; st > 0; st >>= 1) {
        if ((int)threadIdx.x < st) {
            rs1[threadIdx.x] += rs1[threadIdx.x + st];
            rs2[threadIdx.x] += rs2[threadIdx.x + st];
        }
        __syncthreads();
    }
    const float mu  = rs1[0] * (1.0f / 1024.0f);
    const float var = rs2[0] * (1.0f / 1024.0f) - mu * mu;
    const float rrs = rsqrtf(var + 1e-5f);
#pragma unroll
    for (int i = 0; i < 4; ++i) {
        int c = threadIdx.x + 256 * i;
        y[(long)row * 1024 + c] = f2bf((v[i] - mu) * rrs * g[c] + b[c]);
    }
}

// ---------------------------------------------------------------------------
// Tiled bf16 WMMA GEMM:  Out(M,N) = A(M,K) @ B(N,K)^T  + epilogue
//   EPI 0: +bias              -> bf16 out
//   EPI 1: +bias +resid       -> f32 out
//   EPI 2: relu((+bias)*mask) -> bf16 out
//   EPI 3: resid+(+bias)*mask -> f32 out
// Block tile 128x128, K-tile 32, 256 threads = 8 wave32 waves (4 x 2),
// each wave computes a 32x64 sub-tile = 2x4 WMMA accumulators.
// A/B tiles are staged into LDS with GLOBAL_LOAD_ASYNC_TO_LDS_B128 (ASYNCcnt).
// ---------------------------------------------------------------------------
template <int EPI>
__global__ __launch_bounds__(256)
void gemm_bf16_kernel(const unsigned short* __restrict__ A,
                      const unsigned short* __restrict__ Bw,
                      const float* __restrict__ bias,
                      const float* __restrict__ resid,
                      const float* __restrict__ mask,
                      float* __restrict__ outF,
                      unsigned short* __restrict__ outH,
                      int M, int N, int K) {
    __shared__ __align__(32) unsigned short As[128 * 32];
    __shared__ __align__(32) unsigned short Bs[128 * 32];

    const int tid  = threadIdx.x;
    const int lane = tid & 31;
    const int w    = tid >> 5;
    const int wm   = w & 3;     // 0..3  -> 32 rows each
    const int wn   = w >> 2;    // 0..1  -> 64 cols each
    const int lm   = lane & 15;
    const int kh   = (lane >> 4) * 16;   // K half select

    const long rowBase = (long)blockIdx.y * 128;
    const long colBase = (long)blockIdx.x * 128;

    v8f acc[2][4];
#pragma unroll
    for (int mt = 0; mt < 2; ++mt)
#pragma unroll
        for (int nt = 0; nt < 4; ++nt)
            acc[mt][nt] = (v8f){0.f, 0.f, 0.f, 0.f, 0.f, 0.f, 0.f, 0.f};

    const int lr = tid >> 1;          // 0..127 : tile row loaded by this thread
    const int lc = (tid & 1) * 16;    // 0 / 16 : K offset

    for (int kt = 0; kt < K; kt += 32) {
        const unsigned short* gA = A  + (rowBase + lr) * (long)K + kt + lc;
        const unsigned short* gB = Bw + (colBase + lr) * (long)K + kt + lc;
        __syncthreads();   // previous tile fully consumed before LDS overwrite
        // async global -> LDS staging (ASYNCcnt)
        async_g2l_b128(As + lr * 32 + lc,     gA);
        async_g2l_b128(As + lr * 32 + lc + 8, gA + 8);
        async_g2l_b128(Bs + lr * 32 + lc,     gB);
        async_g2l_b128(Bs + lr * 32 + lc + 8, gB + 8);
        if (kt + 32 < K) {   // prefetch next K-tile (global_prefetch_b8)
            __builtin_prefetch((const void*)(gA + 32), 0, 1);
            __builtin_prefetch((const void*)(gB + 32), 0, 1);
        }
        wait_async0();     // own async copies landed in LDS
        __syncthreads();   // everyone's copies visible

        v8u af[2], bfr[4];
#pragma unroll
        for (int mt = 0; mt < 2; ++mt)
            af[mt] = *(const v8u*)(As + (wm * 32 + mt * 16 + lm) * 32 + kh);
#pragma unroll
        for (int nt = 0; nt < 4; ++nt)
            bfr[nt] = *(const v8u*)(Bs + (wn * 64 + nt * 16 + lm) * 32 + kh);
#pragma unroll
        for (int mt = 0; mt < 2; ++mt)
#pragma unroll
            for (int nt = 0; nt < 4; ++nt)
                acc[mt][nt] = wmma_bf16(af[mt], bfr[nt], acc[mt][nt]);
    }

    // epilogue
#pragma unroll
    for (int mt = 0; mt < 2; ++mt) {
#pragma unroll
        for (int nt = 0; nt < 4; ++nt) {
            const long col = colBase + wn * 64 + nt * 16 + lm;
            const float bb = bias[col];
#pragma unroll
            for (int r = 0; r < 8; ++r) {
                const long row = rowBase + wm * 32 + mt * 16 + r + 8 * (lane >> 4);
                const long o   = row * (long)N + col;
                float v = acc[mt][nt][r] + bb;
                if (EPI == 0) {
                    outH[o] = f2bf(v);
                } else if (EPI == 1) {
                    outF[o] = resid[o] + v;
                } else if (EPI == 2) {
                    v = (mask[o] > 0.5f) ? fmaxf(v, 0.f) : 0.f;
                    outH[o] = f2bf(v);
                } else {
                    v = (mask[o] > 0.5f) ? v : 0.f;
                    outF[o] = resid[o] + v;
                }
            }
        }
    }
}

// ---------------------------------------------------------------------------
// Fused sparse attention for one (b, h, 32-row tile).
// Phase 1: S = (Q K^T) * scale via WMMA straight from global qkv -> LDS (32x1024 f32)
// Phase 2: per-row top-k (k=102) threshold via binary search + masked softmax
// Phase 3: O = P V via WMMA (P converted f32->bf16 from LDS)
// qkv layout: (B*N, 3072) bf16, [q | k | v], head-major inside each third.
// ---------------------------------------------------------------------------
__global__ __launch_bounds__(256)
void attn_kernel(const unsigned short* __restrict__ qkv,
                 unsigned short* __restrict__ ao) {
    extern __shared__ float S[];          // 32 * 1024 floats = 128 KB
    const int rb   = blockIdx.x;          // row-block 0..31
    const int bh   = blockIdx.y;          // 0..127
    const int bb   = bh >> 4;
    const int hh   = bh & 15;
    const int tid  = threadIdx.x;
    const int lane = tid & 31;
    const int w    = tid >> 5;
    const int lm   = lane & 15;
    const int kh   = (lane >> 4) * 16;

    // ---- Phase 1: scores -------------------------------------------------
    v8u aq[2][2];
#pragma unroll
    for (int mt = 0; mt < 2; ++mt)
#pragma unroll
        for (int ks = 0; ks < 2; ++ks) {
            const long row = (long)bb * 1024 + rb * 32 + mt * 16 + lm;
            aq[mt][ks] = *(const v8u*)(qkv + row * 3072 + hh * 64 + ks * 32 + kh);
        }

    for (int nt8 = 0; nt8 < 8; ++nt8) {
        const int ncol0 = (w * 8 + nt8) * 16;
        v8f acc0 = (v8f){0.f, 0.f, 0.f, 0.f, 0.f, 0.f, 0.f, 0.f};
        v8f acc1 = acc0;
#pragma unroll
        for (int ks = 0; ks < 2; ++ks) {
            const long krow = (long)bb * 1024 + ncol0 + lm;
            v8u bk = *(const v8u*)(qkv + krow * 3072 + 1024 + hh * 64 + ks * 32 + kh);
            acc0 = wmma_bf16(aq[0][ks], bk, acc0);
            acc1 = wmma_bf16(aq[1][ks], bk, acc1);
        }
        const int col = ncol0 + lm;
#pragma unroll
        for (int r = 0; r < 8; ++r) {
            const int r0 = r + 8 * (lane >> 4);
            S[r0 * 1024 + col]        = acc0[r] * 0.125f;   // scale = hd^-0.5
            S[(16 + r0) * 1024 + col] = acc1[r] * 0.125f;
        }
    }
    __syncthreads();

    // ---- Phase 2: top-k threshold + softmax ------------------------------
    for (int rr = 0; rr < 4; ++rr) {
        const int row = w * 4 + rr;
        float* Sr = S + row * 1024;
        float v[32];
        float mx = -3.4e38f, mn = 3.4e38f;
#pragma unroll
        for (int i = 0; i < 32; ++i) {
            v[i] = Sr[i * 32 + lane];
            mx = fmaxf(mx, v[i]);
            mn = fminf(mn, v[i]);
        }
        for (int m = 16; m >= 1; m >>= 1) {
            mx = fmaxf(mx, __shfl_xor(mx, m, 32));
            mn = fminf(mn, __shfl_xor(mn, m, 32));
        }
        float lo = mn, hi = mx;
        for (int it = 0; it < 24; ++it) {
            const float t = 0.5f * (lo + hi);
            float c = 0.f;
#pragma unroll
            for (int i = 0; i < 32; ++i) c += (v[i] >= t) ? 1.f : 0.f;
            for (int m = 16; m >= 1; m >>= 1) c += __shfl_xor(c, m, 32);
            if (c >= 102.f) lo = t; else hi = t;   // k_sparse = int(0.1*1024)
        }
        const float thr = lo;
        float e[32];
        float s = 0.f;
#pragma unroll
        for (int i = 0; i < 32; ++i) {
            e[i] = (v[i] >= thr) ? __expf(v[i] - mx) : 0.f;
            s += e[i];
        }
        for (int m = 16; m >= 1; m >>= 1) s += __shfl_xor(s, m, 32);
        const float inv = 1.f / s;
#pragma unroll
        for (int i = 0; i < 32; ++i) Sr[i * 32 + lane] = e[i] * inv;
    }
    __syncthreads();

    // ---- Phase 3: O = P @ V ---------------------------------------------
    const int mt = w >> 2;   // 0..1
    const int nt = w & 3;    // 0..3
    v8f acc = (v8f){0.f, 0.f, 0.f, 0.f, 0.f, 0.f, 0.f, 0.f};
    const int dcol = nt * 16 + lm;
    for (int ks = 0; ks < 32; ++ks) {
        // A fragment: probabilities from LDS, f32 -> bf16
        const float* sp = S + (mt * 16 + lm) * 1024 + ks * 32 + kh;
        v8u af;
#pragma unroll
        for (int j = 0; j < 8; ++j)
            af[j] = (unsigned int)f2bf(sp[2 * j]) |
                    ((unsigned int)f2bf(sp[2 * j + 1]) << 16);
        // B fragment: V column `dcol`, 16 K values (strided over tokens)
        v8u bfv;
#pragma unroll
        for (int j = 0; j < 8; ++j) {
            const long t0 = (long)bb * 1024 + ks * 32 + kh + 2 * j;
            const unsigned int u0 = qkv[t0 * 3072 + 2048 + hh * 64 + dcol];
            const unsigned int u1 = qkv[(t0 + 1) * 3072 + 2048 + hh * 64 + dcol];
            bfv[j] = u0 | (u1 << 16);
        }
        acc = wmma_bf16(af, bfv, acc);
    }
#pragma unroll
    for (int r = 0; r < 8; ++r) {
        const int  r0  = r + 8 * (lane >> 4);
        const long row = (long)bb * 1024 + rb * 32 + mt * 16 + r0;
        ao[row * 1024 + hh * 64 + dcol] = f2bf(acc[r]);
    }
}

// ---------------------------------------------------------------------------
// Host launcher
// ---------------------------------------------------------------------------
extern "C" void kernel_launch(void* const* d_in, const int* in_sizes, int n_in,
                              void* d_out, int out_size, void* d_ws, size_t ws_size,
                              hipStream_t stream) {
    (void)in_sizes; (void)n_in; (void)out_size; (void)ws_size;
    const float* x      = (const float*)d_in[0];
    const float* ln1_w  = (const float*)d_in[1];
    const float* ln1_b  = (const float*)d_in[2];
    const float* qkv_w  = (const float*)d_in[3];
    const float* qkv_b  = (const float*)d_in[4];
    const float* out_w  = (const float*)d_in[5];
    const float* out_b  = (const float*)d_in[6];
    const float* ln2_w  = (const float*)d_in[7];
    const float* ln2_b  = (const float*)d_in[8];
    const float* w1     = (const float*)d_in[9];
    const float* b1     = (const float*)d_in[10];
    const float* w2     = (const float*)d_in[11];
    const float* b2     = (const float*)d_in[12];
    const float* mask1  = (const float*)d_in[13];
    const float* mask2  = (const float*)d_in[14];
    float*       out    = (float*)d_out;

    // workspace carve (256B aligned)
    char* ws = (char*)d_ws;
    auto carve = [&](size_t bytes) -> char* {
        char* p = ws;
        ws += (bytes + 255) & ~(size_t)255;
        return p;
    };
    unsigned short* qkvw_h = (unsigned short*)carve(3072UL * 1024 * 2);
    unsigned short* outw_h = (unsigned short*)carve(1024UL * 1024 * 2);
    unsigned short* w1_h   = (unsigned short*)carve(4096UL * 1024 * 2);
    unsigned short* w2_h   = (unsigned short*)carve(1024UL * 4096 * 2);
    unsigned short* xn_h   = (unsigned short*)carve(8192UL * 1024 * 2);  // reused for LN2
    unsigned short* qkv_h  = (unsigned short*)carve(8192UL * 3072 * 2);
    unsigned short* ao_h   = (unsigned short*)carve(8192UL * 1024 * 2);
    float*          x1_f   = (float*)carve(8192UL * 1024 * 4);
    unsigned short* h1_h   = (unsigned short*)carve(8192UL * 4096 * 2);

    // weights fp32 -> bf16
    cvt_f32_bf16<<<2048, 256, 0, stream>>>(qkv_w, qkvw_h, 3072L * 1024);
    cvt_f32_bf16<<<1024, 256, 0, stream>>>(out_w, outw_h, 1024L * 1024);
    cvt_f32_bf16<<<2048, 256, 0, stream>>>(w1,    w1_h,   4096L * 1024);
    cvt_f32_bf16<<<2048, 256, 0, stream>>>(w2,    w2_h,   1024L * 4096);

    // LN1
    ln_bf16_kernel<<<8192, 256, 0, stream>>>(x, ln1_w, ln1_b, xn_h);

    // QKV projection: (8192,1024) @ (3072,1024)^T
    gemm_bf16_kernel<0><<<dim3(24, 64), 256, 0, stream>>>(
        xn_h, qkvw_h, qkv_b, nullptr, nullptr, nullptr, qkv_h, 8192, 3072, 1024);

    // fused sparse attention (dynamic LDS 128 KB)
    hipFuncSetAttribute((const void*)attn_kernel,
                        hipFuncAttributeMaxDynamicSharedMemorySize, 131072);
    attn_kernel<<<dim3(32, 128), 256, 131072, stream>>>(qkv_h, ao_h);

    // out projection + residual -> x1
    gemm_bf16_kernel<1><<<dim3(8, 64), 256, 0, stream>>>(
        ao_h, outw_h, out_b, x, nullptr, x1_f, nullptr, 8192, 1024, 1024);

    // LN2
    ln_bf16_kernel<<<8192, 256, 0, stream>>>(x1_f, ln2_w, ln2_b, xn_h);

    // FFN1: relu((h @ w1^T + b1) * m1) -> bf16
    gemm_bf16_kernel<2><<<dim3(32, 64), 256, 0, stream>>>(
        xn_h, w1_h, b1, nullptr, mask1, nullptr, h1_h, 8192, 4096, 1024);

    // FFN2: x1 + (h @ w2^T + b2) * m2 -> out
    gemm_bf16_kernel<3><<<dim3(8, 64), 256, 0, stream>>>(
        h1_h, w2_h, b2, x1_f, mask2, out, nullptr, 8192, 1024, 4096);
}
